// GTG_85761906967201
// MI455X (gfx1250) — matrix operand order repaired and checked by reading, not compiled, not run on hardware
//
#include <hip/hip_runtime.h>
#include <stdint.h>

// ---------------------------------------------------------------------------
// GTG (graph transduction game) for MI455X / gfx1250, wave32 + WMMA bf16.
//   W  = relu( corr(fc7) )            [4096 x 4096] fp32  (output #2)
//   ps = replicator(W, ps0, 5 iters)  [4096 x 1000] fp32  (output #1)
// GEMMs run as bf16 WMMA with fp32 accumulate; hi/lo bf16 splitting recovers
// ~fp32 accuracy (drops only the lo*lo term). The Gram GEMM uses CDNA5
// async global->LDS copies (ASYNCcnt) with double-buffered LDS.
// ---------------------------------------------------------------------------

typedef __attribute__((ext_vector_type(16))) __bf16 v16bf;
typedef __attribute__((ext_vector_type(8)))  __bf16 bf16x8;
typedef __attribute__((ext_vector_type(8)))  float  v8f;
typedef __attribute__((ext_vector_type(4)))  float  f32x4;

#define GTG_N     4096
#define GTG_D     2048
#define GTG_M     1000
#define GTG_MPAD  1024
#define GTG_NL    2048
#define GTG_ITERS 5

union FragBF {
  v16bf  v;
  bf16x8 h[2];
};

__device__ __forceinline__ v8f wmma_bf16f32(const FragBF& a, const FragBF& b, v8f c) {
  // (neg_a, A, neg_b, B, c_mod, C, reuse_a, reuse_b)
  return __builtin_amdgcn_wmma_f32_16x16x32_bf16(false, a.v, false, b.v,
                                                 (short)0, c, false, false);
}

// CDNA5 async copy: 16B per lane, global -> LDS, tracked by ASYNCcnt.
__device__ __forceinline__ void async_b128(void* lds, const void* g) {
  unsigned l = (unsigned)(uintptr_t)lds;        // low 32 bits = LDS offset
  uint64_t a = (uint64_t)(uintptr_t)g;
  asm volatile("global_load_async_to_lds_b128 %0, %1, off"
               :: "v"(l), "v"(a) : "memory");
}

__device__ __forceinline__ void wait_async0() {
  asm volatile("s_wait_asynccnt 0x0" ::: "memory");
}

// ---------------------------------------------------------------------------
// 1) Center rows of fc7, produce x_hi/x_lo (bf16 split) and 1/||x||.
// ---------------------------------------------------------------------------
__global__ void gtg_center(const float* __restrict__ fc7,
                           __bf16* __restrict__ xhi,
                           __bf16* __restrict__ xlo,
                           float*  __restrict__ inv_norm) {
  __shared__ float red[256];
  const int row = blockIdx.x;
  const int tid = threadIdx.x;
  const float* r = fc7 + (size_t)row * GTG_D;

  float s = 0.0f;
  for (int j = tid; j < GTG_D; j += 256) s += r[j];
  red[tid] = s;
  __syncthreads();
  for (int o = 128; o > 0; o >>= 1) {
    if (tid < o) red[tid] += red[tid + o];
    __syncthreads();
  }
  const float mean = red[0] * (1.0f / (float)GTG_D);
  __syncthreads();

  float ss = 0.0f;
  for (int j = tid; j < GTG_D; j += 256) {
    float v = r[j] - mean;
    __bf16 hi = (__bf16)v;
    __bf16 lo = (__bf16)(v - (float)hi);
    xhi[(size_t)row * GTG_D + j] = hi;
    xlo[(size_t)row * GTG_D + j] = lo;
    ss += v * v;
  }
  red[tid] = ss;
  __syncthreads();
  for (int o = 128; o > 0; o >>= 1) {
    if (tid < o) red[tid] += red[tid + o];
    __syncthreads();
  }
  if (tid == 0) inv_norm[row] = rsqrtf(red[0]);
}

// ---------------------------------------------------------------------------
// 2) Gram GEMM: W = relu( (x @ x^T) * inv_i * inv_j ), split bf16 WMMA.
//    Block tile 128(M) x 64(N), K-step 32, 8 waves, wave tile 32x32.
//    Double-buffered LDS filled by global_load_async_to_lds_b128: the async
//    engine fills buffer bi^1 for K+32 while WMMAs consume buffer bi.
// ---------------------------------------------------------------------------
#define LDSP 40   // 32 + 8 bf16 padding: 80-byte row stride, 16B aligned

__global__ void gtg_gram(const __bf16* __restrict__ xhi,
                         const __bf16* __restrict__ xlo,
                         const float*  __restrict__ inv_norm,
                         float* __restrict__ W) {
  __shared__ __bf16 Ah[2][128][LDSP];
  __shared__ __bf16 Al[2][128][LDSP];
  __shared__ __bf16 Bh[2][64][LDSP];
  __shared__ __bf16 Bl[2][64][LDSP];

  const int tid  = threadIdx.x;
  const int lane = tid & 31;
  const int wave = tid >> 5;
  const int wm   = (wave & 3) * 32;   // wave M offset in block tile
  const int wn   = (wave >> 2) * 32;  // wave N offset in block tile
  const int l15  = lane & 15;
  const int lhi  = lane >> 4;

  const int rowA = blockIdx.y * 128;  // M block (rows of x)
  const int rowB = blockIdx.x * 64;   // N block (rows of x, used as B columns)

  const int la = tid >> 2;          // 0..63
  const int lc = (tid & 3) * 8;     // 0,8,16,24

  v8f acc[2][2];
#pragma unroll
  for (int i = 0; i < 2; ++i)
#pragma unroll
    for (int j = 0; j < 2; ++j) acc[i][j] = {};

  // issue async stage for one K-tile into buffer bi
  auto issue_stage = [&](int k0, int bi) {
#pragma unroll
    for (int p = 0; p < 2; ++p) {
      const int r = la + p * 64;
      const size_t g = (size_t)(rowA + r) * GTG_D + k0 + lc;
      async_b128(&Ah[bi][r][lc], xhi + g);
      async_b128(&Al[bi][r][lc], xlo + g);
    }
    const size_t g = (size_t)(rowB + la) * GTG_D + k0 + lc;
    async_b128(&Bh[bi][la][lc], xhi + g);
    async_b128(&Bl[bi][la][lc], xlo + g);
  };

  issue_stage(0, 0);
  int bi = 0;

  for (int k0 = 0; k0 < GTG_D; k0 += 32) {
    wait_async0();       // this wave's async fills of buffer bi are done
    __syncthreads();     // all waves' fills done; prev-step LDS reads drained

    if (k0 + 32 < GTG_D) issue_stage(k0 + 32, bi ^ 1);  // overlap with WMMAs

    // ---- fragments ----
    FragBF aH[2], aL[2], bH[2], bL[2];
    const int akh = lhi * 8;
#pragma unroll
    for (int i = 0; i < 2; ++i) {
      const int r = wm + i * 16 + l15;
      aH[i].h[0] = *(const bf16x8*)&Ah[bi][r][akh];
      aH[i].h[1] = *(const bf16x8*)&Ah[bi][r][16 + akh];
      aL[i].h[0] = *(const bf16x8*)&Al[bi][r][akh];
      aL[i].h[1] = *(const bf16x8*)&Al[bi][r][16 + akh];
    }
    const int bks = lhi * 8;
#pragma unroll
    for (int j = 0; j < 2; ++j) {
      const int r = wn + j * 16 + l15;
      bH[j].h[0] = *(const bf16x8*)&Bh[bi][r][bks];
      bH[j].h[1] = *(const bf16x8*)&Bh[bi][r][16 + bks];
      bL[j].h[0] = *(const bf16x8*)&Bl[bi][r][bks];
      bL[j].h[1] = *(const bf16x8*)&Bl[bi][r][16 + bks];
    }

    // ---- x.x^T ≈ hi.hi + hi.lo + lo.hi ----
#pragma unroll
    for (int i = 0; i < 2; ++i)
#pragma unroll
      for (int j = 0; j < 2; ++j) {
        acc[i][j] = wmma_bf16f32(aH[i], bH[j], acc[i][j]);
        acc[i][j] = wmma_bf16f32(aH[i], bL[j], acc[i][j]);
        acc[i][j] = wmma_bf16f32(aL[i], bH[j], acc[i][j]);
      }

    bi ^= 1;
  }

  // ---- epilogue: scale by inv_norm outer product, ReLU, store fp32 ----
#pragma unroll
  for (int i = 0; i < 2; ++i) {
#pragma unroll
    for (int j = 0; j < 2; ++j) {
      const int gn = rowB + wn + j * 16 + l15;
      const float sj = inv_norm[gn];
#pragma unroll
      for (int r = 0; r < 8; ++r) {
        const int gm = rowA + wm + i * 16 + lhi * 8 + r;
        float v = acc[i][j][r] * inv_norm[gm] * sj;
        W[(size_t)gm * GTG_N + gn] = v > 0.0f ? v : 0.0f;
      }
    }
  }
}

// ---------------------------------------------------------------------------
// 3) ps init: labeled rows one-hot, unlabeled rows = prior; zero-pad to 1024.
// ---------------------------------------------------------------------------
__global__ void gtg_init_ps(const float* __restrict__ probs,
                            const int* __restrict__ labs,
                            float* __restrict__ p) {
  const int row = blockIdx.x;
  const int tid = threadIdx.x;
  if (row < GTG_NL) {
    const int lab = labs[row];
    for (int j = tid; j < GTG_MPAD; j += 256)
      p[(size_t)row * GTG_MPAD + j] = (j == lab) ? 1.0f : 0.0f;
  } else {
    for (int j = tid; j < GTG_MPAD; j += 256)
      p[(size_t)row * GTG_MPAD + j] =
          (j < GTG_M) ? probs[(size_t)row * GTG_M + j] : 0.0f;
  }
}

// ---------------------------------------------------------------------------
// 4) t = W @ p.  W fp32 (from d_out) split hi/lo bf16 while staging; p fp32
//    transposed through LDS into B-fragment order. Block 128x64, K-step 32.
//    (Synchronous staging: the A tile needs an in-flight f32->bf16 hi/lo
//    conversion and the B tile a transpose, which async copies cannot do.)
// ---------------------------------------------------------------------------
__global__ void gtg_gemm_wp(const float* __restrict__ W,
                            const float* __restrict__ p,
                            float* __restrict__ t) {
  __shared__ __bf16 Ah[128][LDSP];
  __shared__ __bf16 Al[128][LDSP];
  __shared__ __bf16 Bt[64][LDSP];   // Bt[n][k] = p[k0+k][n0+n]

  const int tid  = threadIdx.x;
  const int lane = tid & 31;
  const int wave = tid >> 5;
  const int wm   = (wave & 3) * 32;
  const int wn   = (wave >> 2) * 32;
  const int l15  = lane & 15;
  const int lhi  = lane >> 4;

  const int rowA = blockIdx.y * 128;   // M block of W
  const int n0   = blockIdx.x * 64;    // N block of p columns

  v8f acc[2][2];
#pragma unroll
  for (int i = 0; i < 2; ++i)
#pragma unroll
    for (int j = 0; j < 2; ++j) acc[i][j] = {};

  for (int k0 = 0; k0 < GTG_N; k0 += 32) {
    // ---- stage W tile, hi/lo split ----
#pragma unroll
    for (int ps = 0; ps < 4; ++ps) {
      const int r = (tid >> 3) + ps * 32;     // 0..127
      const int c = (tid & 7) * 4;            // 0..28
      const size_t g = (size_t)(rowA + r) * GTG_N + k0 + c;
      f32x4 w4 = *(const f32x4*)(W + g);
      if (k0 + 32 < GTG_N) __builtin_prefetch(W + g + 32, 0, 1);
#pragma unroll
      for (int e = 0; e < 4; ++e) {
        __bf16 hi = (__bf16)w4[e];
        Ah[r][c + e] = hi;
        Al[r][c + e] = (__bf16)(w4[e] - (float)hi);
      }
    }
    // ---- stage p tile transposed ----
    {
      const int k = tid >> 3;          // 0..31
      const int n = (tid & 7) * 8;     // 0..56
      const float* src = p + (size_t)(k0 + k) * GTG_MPAD + n0 + n;
#pragma unroll
      for (int e = 0; e < 8; ++e) Bt[n + e][k] = (__bf16)src[e];
    }
    __syncthreads();

    FragBF aH[2], aL[2], bF[2];
    const int akh = lhi * 8;
#pragma unroll
    for (int i = 0; i < 2; ++i) {
      const int r = wm + i * 16 + l15;
      aH[i].h[0] = *(const bf16x8*)&Ah[r][akh];
      aH[i].h[1] = *(const bf16x8*)&Ah[r][16 + akh];
      aL[i].h[0] = *(const bf16x8*)&Al[r][akh];
      aL[i].h[1] = *(const bf16x8*)&Al[r][16 + akh];
    }
    const int bks = lhi * 8;
#pragma unroll
    for (int j = 0; j < 2; ++j) {
      const int r = wn + j * 16 + l15;
      bF[j].h[0] = *(const bf16x8*)&Bt[r][bks];
      bF[j].h[1] = *(const bf16x8*)&Bt[r][16 + bks];
    }

#pragma unroll
    for (int i = 0; i < 2; ++i)
#pragma unroll
      for (int j = 0; j < 2; ++j) {
        acc[i][j] = wmma_bf16f32(aH[i], bF[j], acc[i][j]);
        acc[i][j] = wmma_bf16f32(aL[i], bF[j], acc[i][j]);
      }
    __syncthreads();
  }

#pragma unroll
  for (int i = 0; i < 2; ++i)
#pragma unroll
    for (int j = 0; j < 2; ++j) {
      const int gn = n0 + wn + j * 16 + l15;
#pragma unroll
      for (int r = 0; r < 8; ++r) {
        const int gm = rowA + wm + i * 16 + lhi * 8 + r;
        t[(size_t)gm * GTG_MPAD + gn] = acc[i][j][r];
      }
    }
}

// ---------------------------------------------------------------------------
// 5) Replicator update: q = p .* t ; p = q / rowsum(q)   (in place, padded).
// ---------------------------------------------------------------------------
__global__ void gtg_update(const float* __restrict__ t, float* __restrict__ p) {
  __shared__ float red[256];
  const int row = blockIdx.x;
  const int tid = threadIdx.x;
  const size_t base = (size_t)row * GTG_MPAD;

  float s = 0.0f;
  for (int j = tid; j < GTG_M; j += 256) s += p[base + j] * t[base + j];
  red[tid] = s;
  __syncthreads();
  for (int o = 128; o > 0; o >>= 1) {
    if (tid < o) red[tid] += red[tid + o];
    __syncthreads();
  }
  const float inv = 1.0f / red[0];
  for (int j = tid; j < GTG_MPAD; j += 256) {
    float q = (j < GTG_M) ? p[base + j] * t[base + j] * inv : 0.0f;
    p[base + j] = q;
  }
}

// ---------------------------------------------------------------------------
// 6) Strip padding into the output buffer.
// ---------------------------------------------------------------------------
__global__ void gtg_copy_ps(const float* __restrict__ p, float* __restrict__ out) {
  const int row = blockIdx.x;
  for (int j = threadIdx.x; j < GTG_M; j += 256)
    out[(size_t)row * GTG_M + j] = p[(size_t)row * GTG_MPAD + j];
}

// ---------------------------------------------------------------------------
extern "C" void kernel_launch(void* const* d_in, const int* in_sizes, int n_in,
                              void* d_out, int out_size, void* d_ws, size_t ws_size,
                              hipStream_t stream) {
  (void)in_sizes; (void)n_in; (void)out_size; (void)ws_size;

  const float* fc7   = (const float*)d_in[0];
  const float* probs = (const float*)d_in[1];
  const int*   labs  = (const int*)d_in[2];
  // d_in[3] = L, d_in[4] = U: implicit (L = [0,2048), U = [2048,4096))

  float* ps_out = (float*)d_out;                         // [4096 x 1000]
  float* Wout   = ps_out + (size_t)GTG_N * GTG_M;        // [4096 x 4096]

  // Workspace layout (~67 MB)
  char* ws = (char*)d_ws;
  __bf16* xhi = (__bf16*)ws;                                    // 16.8 MB
  __bf16* xlo = xhi + (size_t)GTG_N * GTG_D;                    // 16.8 MB
  float* inv_norm = (float*)(xlo + (size_t)GTG_N * GTG_D);      // 16 KB
  float* p = inv_norm + GTG_N;                                  // 16.8 MB
  float* t = p + (size_t)GTG_N * GTG_MPAD;                      // 16.8 MB

  gtg_center<<<GTG_N, 256, 0, stream>>>(fc7, xhi, xlo, inv_norm);

  gtg_gram<<<dim3(GTG_N / 64, GTG_N / 128), 256, 0, stream>>>(xhi, xlo, inv_norm, Wout);

  gtg_init_ps<<<GTG_N, 256, 0, stream>>>(probs, labs, p);

  for (int it = 0; it < GTG_ITERS; ++it) {
    gtg_gemm_wp<<<dim3(GTG_MPAD / 64, GTG_N / 128), 256, 0, stream>>>(Wout, p, t);
    gtg_update<<<GTG_N, 256, 0, stream>>>(t, p);
  }

  gtg_copy_ps<<<GTG_N, 256, 0, stream>>>(p, ps_out);
}
